// LSTMModel_78812649881962
// MI455X (gfx1250) — matrix operand order repaired and checked by reading, not compiled, not run on hardware
//
#include <hip/hip_runtime.h>

// =====================================================================
// 2-layer LSTM (B=512,T=1024,D=6,H=128) + per-timestep MLP decoder.
// Persistent-workgroup design: batch is the only parallel axis during the
// recurrence, so 16 workgroups each own 32 batch rows and run all 1024
// timesteps with the layer's weights resident in LDS (f16, WMMA B layout).
// All GEMM work goes through v_wmma_f32_16x16x32_f16 with f32 accumulators;
// cell state c lives in VGPRs across timesteps. Layer-1 kernel fuses the
// decoder MLP so h2 never touches HBM. h1 crosses HBM once, in f16.
// =====================================================================

#define DEVINL __device__ __forceinline__

typedef __attribute__((ext_vector_type(16))) _Float16 v16h;
typedef __attribute__((ext_vector_type(8)))  _Float16 v8h;
typedef __attribute__((ext_vector_type(8)))  float    v8f;

// ---- problem sizes ----
constexpr int B_ = 512, T_ = 1024, D_ = 6, H_ = 128, G_ = 4 * H_;  // G = 4H = 512
constexpr int BM   = 32;           // batch rows per workgroup
constexpr int NWG  = B_ / BM;      // 16 workgroups
constexpr int NTHR = 256;          // 8 wave32 per workgroup
constexpr int NT_G = G_ / 16;      // 32 gate n-tiles
constexpr int KT_H = H_ / 32;      // 4 k-tiles across hidden dim
constexpr int FRAG = 512;          // f16 elements per packed 32x16 B fragment
constexpr int HRS  = H_ + 8;       // padded LDS row stride (f16) for h tiles
constexpr int XRS0 = 40;           // padded row stride for layer-0 x tile (K=6->32)
constexpr int YRS  = 72;           // padded row stride for decoder y tile

// ---- workspace layout (bytes) ----
constexpr size_t WS_WIH0 = 0;                                         // 32 KB
constexpr size_t WS_WHH0 = WS_WIH0 + (size_t)1   * NT_G * FRAG * 2;   // 128 KB
constexpr size_t WS_WIH1 = WS_WHH0 + (size_t)KT_H * NT_G * FRAG * 2;  // 128 KB
constexpr size_t WS_WHH1 = WS_WIH1 + (size_t)KT_H * NT_G * FRAG * 2;  // 128 KB
constexpr size_t WS_WD1  = WS_WHH1 + (size_t)KT_H * NT_G * FRAG * 2;  // 16 KB
constexpr size_t WS_B0   = WS_WD1 + (size_t)KT_H * 4 * FRAG * 2;
constexpr size_t WS_B1   = WS_B0 + 512 * 4;
constexpr size_t WS_BD1  = WS_B1 + 512 * 4;
constexpr size_t WS_MISC = WS_BD1 + 64 * 4;       // 80 floats: [0..63]=Wd2, [64]=bd2
constexpr size_t WS_H1   = (size_t)1 << 19;       // h1 f16 [B][T][H] = 128 MB

constexpr size_t lds_bytes(bool L1) {
  size_t s = 0;
  s += (size_t)(L1 ? KT_H : 1) * NT_G * FRAG * 2;  // packed W_ih fragments
  s += (size_t)KT_H * NT_G * FRAG * 2;             // packed W_hh fragments
  s += 512 * 4;                                    // combined gate bias (f32)
  s += (size_t)BM * (L1 ? HRS : XRS0) * 2;         // x_t staging tile
  s += (size_t)2 * BM * HRS * 2;                   // h ping-pong tiles
  if (L1) {
    s += (size_t)KT_H * 4 * FRAG * 2;              // packed Wd1 fragments
    s += 64 * 4;                                   // bd1
    s += 80 * 4;                                   // Wd2 + bd2
    s += (size_t)BM * YRS * 2;                     // y tile
  }
  return s;
}
static_assert(lds_bytes(true)  <= 320u * 1024u, "layer-1 LDS over budget");
static_assert(lds_bytes(false) <= 320u * 1024u, "layer-0 LDS over budget");

// ---- helpers ----
DEVINL void cp16(void* dst, const void* src, int bytes, int tid, int nthr) {
  const uint4* s = (const uint4*)src;
  uint4* d = (uint4*)dst;
  for (int i = tid; i < bytes / 16; i += nthr) d[i] = s[i];
}

// A fragment (16x32 f16): lane&15 = row m; lanes>=16 take k+8; two b128 loads.
DEVINL v16h load_A(const _Float16* base, int row0, int rs, int k0, int lane) {
  const _Float16* p = base + (size_t)(row0 + (lane & 15)) * rs + k0 + ((lane >> 4) << 3);
  v8h lo = *(const v8h*)p;
  v8h hi = *(const v8h*)(p + 16);
  v16h a;
#pragma unroll
  for (int s = 0; s < 8; ++s) { a[s] = lo[s]; a[s + 8] = hi[s]; }
  return a;
}

// B fragment (32x16 f16): pre-packed so each lane's 16 values are contiguous.
DEVINL v16h load_Bf(const _Float16* frags, int frag, int lane) {
  const _Float16* p = frags + (size_t)frag * FRAG + lane * 16;
  v8h lo = *(const v8h*)p;
  v8h hi = *(const v8h*)(p + 8);
  v16h b;
#pragma unroll
  for (int s = 0; s < 8; ++s) { b[s] = lo[s]; b[s + 8] = hi[s]; }
  return b;
}

DEVINL v8f wmma_f16(v16h a, v16h b, v8f c) {
  return __builtin_amdgcn_wmma_f32_16x16x32_f16(false, a, false, b, (short)0, c,
                                                false, false);
}

DEVINL float sigm(float x) { return 1.0f / (1.0f + __expf(-x)); }
DEVINL float tanh_(float x) {
  x = fminf(fmaxf(x, -15.0f), 15.0f);
  float e = __expf(2.0f * x);
  return (e - 1.0f) / (e + 1.0f);
}

// =====================================================================
// Weight pre-pack: B-matrix fragment layout for V_WMMA_F32_16X16X32_F16.
// Fragment (ko,no): lane L holds B[k = ko*32 + 8*(L/16) + (s&7) + 16*(s>>3)][n = no*16 + L%16], s=0..15,
// stored lane-contiguous. B[k][n] = W[n][k] (we compute x @ W^T).
// =====================================================================
__global__ void pack_wB(const float* __restrict__ W, _Float16* __restrict__ dst,
                        int Nrows, int Kcols, int NT) {
  const int f = blockIdx.x;
  const int lane = threadIdx.x & 31;
  const int ko = f / NT, no = f - ko * NT;
  const int n = no * 16 + (lane & 15);
#pragma unroll
  for (int s = 0; s < 16; ++s) {
    const int k = ko * 32 + ((lane >> 4) << 3) + (s & 7) + ((s >> 3) << 4);
    float v = (k < Kcols && n < Nrows) ? W[(size_t)n * Kcols + k] : 0.0f;
    dst[(size_t)f * FRAG + lane * 16 + s] = (_Float16)v;
  }
}

__global__ void prep_misc(const float* __restrict__ bih0, const float* __restrict__ bhh0,
                          const float* __restrict__ bih1, const float* __restrict__ bhh1,
                          const float* __restrict__ bd1, const float* __restrict__ wd2,
                          const float* __restrict__ bd2, float* __restrict__ o_b0,
                          float* __restrict__ o_b1, float* __restrict__ o_bd1,
                          float* __restrict__ o_misc) {
  const int i = threadIdx.x;  // 512 threads
  o_b0[i] = bih0[i] + bhh0[i];
  o_b1[i] = bih1[i] + bhh1[i];
  if (i < 64) { o_bd1[i] = bd1[i]; o_misc[i] = wd2[i]; }
  if (i == 0) o_misc[64] = bd2[0];
  if (i >= 65 && i < 80) o_misc[i] = 0.0f;
}

// =====================================================================
// Persistent LSTM layer kernel. L1=false: layer 0 (x from fp32 src, K
// padded 6->32, h written to global f16). L1=true: layer 1 (x from f16 h1,
// decoder MLP fused, out written as fp32 [B,T]).
// Wave w owns combos q = 2w, 2w+1: mt = q>>3, nh = q&7 -> the four gate
// tiles (i,f,g,o) of one (m-tile, hidden-tile) pair live in one wave, so
// the gate nonlinearity needs no cross-lane traffic and c stays in VGPRs.
// =====================================================================
template <bool L1>
__global__ __launch_bounds__(NTHR, 1) void lstm_layer(
    const float* __restrict__ src0,      // L0 input [B,T,6] fp32
    const _Float16* __restrict__ xin,    // L1 input [B,T,128] f16
    const _Float16* __restrict__ wxp,    // packed W_ih fragments
    const _Float16* __restrict__ whp,    // packed W_hh fragments
    const float* __restrict__ biasg,     // combined gate bias [512]
    _Float16* __restrict__ hout,         // L0 output [B,T,128] f16
    const _Float16* __restrict__ wd1p,   // packed Wd1 fragments (L1)
    const float* __restrict__ bd1g,      // bd1 [64] (L1)
    const float* __restrict__ miscg,     // Wd2[64], bd2 (L1)
    float* __restrict__ out)             // [B,T] fp32 (L1)
{
  extern __shared__ char smem[];
  char* sp = smem;
  constexpr int KT_X = L1 ? KT_H : 1;
  constexpr int XRS  = L1 ? HRS : XRS0;

  _Float16* s_wx = (_Float16*)sp; sp += (size_t)KT_X * NT_G * FRAG * 2;
  _Float16* s_wh = (_Float16*)sp; sp += (size_t)KT_H * NT_G * FRAG * 2;
  float*    s_b  = (float*)sp;    sp += 512 * 4;
  _Float16* s_x  = (_Float16*)sp; sp += (size_t)BM * XRS * 2;
  _Float16* s_h  = (_Float16*)sp; sp += (size_t)2 * BM * HRS * 2;
  _Float16* s_wd1 = nullptr; float* s_bd1 = nullptr; float* s_misc = nullptr;
  _Float16* s_y = nullptr;
  if constexpr (L1) {
    s_wd1 = (_Float16*)sp; sp += (size_t)KT_H * 4 * FRAG * 2;
    s_bd1 = (float*)sp;    sp += 64 * 4;
    s_misc = (float*)sp;   sp += 80 * 4;
    s_y = (_Float16*)sp;   sp += (size_t)BM * YRS * 2;
  }

  const int tid = threadIdx.x, lane = tid & 31, wave = tid >> 5;
  const int mb0 = blockIdx.x * BM;

  // stage all weights for this layer into LDS (one-time)
  cp16(s_wx, wxp, KT_X * NT_G * FRAG * 2, tid, NTHR);
  cp16(s_wh, whp, KT_H * NT_G * FRAG * 2, tid, NTHR);
  cp16(s_b, biasg, 512 * 4, tid, NTHR);
  if constexpr (L1) {
    cp16(s_wd1, wd1p, KT_H * 4 * FRAG * 2, tid, NTHR);
    cp16(s_bd1, bd1g, 64 * 4, tid, NTHR);
    cp16(s_misc, miscg, 80 * 4, tid, NTHR);
  }
  // zero x staging (layer-0 K-padding relies on this) and h ping-pong (h_{-1}=0)
  for (int i = tid; i < BM * XRS; i += NTHR) s_x[i] = (_Float16)0.0f;
  for (int i = tid; i < 2 * BM * HRS; i += NTHR) s_h[i] = (_Float16)0.0f;
  __syncthreads();

  const int q0 = wave * 2;
  v8f cst[2];  // persistent cell state, f32, one v8f per combo
#pragma unroll
  for (int c = 0; c < 2; ++c)
#pragma unroll
    for (int v = 0; v < 8; ++v) cst[c][v] = 0.0f;

#pragma unroll 1
  for (int t = 0; t < T_; ++t) {
    _Float16* hc = s_h + (size_t)(t & 1) * BM * HRS;        // h_{t-1}
    _Float16* hn = s_h + (size_t)((t + 1) & 1) * BM * HRS;  // h_t

    // ---- stage x_t into LDS ----
    if constexpr (!L1) {
      if (tid < BM * D_) {
        const int r = tid / D_, d = tid - r * D_;
        s_x[r * XRS + d] =
            (_Float16)src0[((size_t)(mb0 + r) * T_ + t) * D_ + d];
      }
    } else {
      for (int i = tid; i < BM * 16; i += NTHR) {
        const int r = i >> 4, c4 = i & 15;
        const uint4* g =
            (const uint4*)(xin + ((size_t)(mb0 + r) * T_ + t) * H_) + c4;
        *((uint4*)(s_x + (size_t)r * XRS) + c4) = *g;
      }
      if (t + 1 < T_) {  // warm next timestep's rows (global_prefetch)
        __builtin_prefetch(
            xin + ((size_t)(mb0 + (tid & 31)) * T_ + (t + 1)) * H_, 0, 0);
      }
    }
    __syncthreads();

    // ---- gates = [x_t, h_{t-1}] @ [W_ih^T; W_hh^T] + bias via WMMA ----
#pragma unroll
    for (int c = 0; c < 2; ++c) {
      const int q = q0 + c, mt = q >> 3, nh = q & 7;
      v16h ax[KT_X], ah[KT_H];
#pragma unroll
      for (int ko = 0; ko < KT_X; ++ko)
        ax[ko] = load_A(s_x, mt * 16, XRS, ko * 32, lane);
#pragma unroll
      for (int ko = 0; ko < KT_H; ++ko)
        ah[ko] = load_A(hc, mt * 16, HRS, ko * 32, lane);

      v8f acc[4];
#pragma unroll
      for (int g = 0; g < 4; ++g) {
        const int nt = g * 8 + nh;
        const float bv = s_b[nt * 16 + (lane & 15)];
        v8f a;
#pragma unroll
        for (int v = 0; v < 8; ++v) a[v] = bv;
#pragma unroll
        for (int ko = 0; ko < KT_X; ++ko)
          a = wmma_f16(ax[ko], load_Bf(s_wx, ko * NT_G + nt, lane), a);
#pragma unroll
        for (int ko = 0; ko < KT_H; ++ko)
          a = wmma_f16(ah[ko], load_Bf(s_wh, ko * NT_G + nt, lane), a);
        acc[g] = a;
      }

      // ---- gate nonlinearity + state update (pure VALU, no shuffles) ----
      const int ncol = nh * 16 + (lane & 15);
      const int mrow = mt * 16 + ((lane >> 4) << 3);
#pragma unroll
      for (int v = 0; v < 8; ++v) {
        const float iv = sigm(acc[0][v]);
        const float fv = sigm(acc[1][v]);
        const float gv = tanh_(acc[2][v]);
        const float ov = sigm(acc[3][v]);
        const float cn = fv * cst[c][v] + iv * gv;
        cst[c][v] = cn;
        hn[(size_t)(mrow + v) * HRS + ncol] = (_Float16)(ov * tanh_(cn));
      }
    }
    __syncthreads();

    if constexpr (!L1) {
      // ---- write h_t tile to global (f16, coalesced b128) ----
      for (int i = tid; i < BM * 16; i += NTHR) {
        const int r = i >> 4, c4 = i & 15;
        *((uint4*)(hout + ((size_t)(mb0 + r) * T_ + t) * H_) + c4) =
            *((const uint4*)(hn + (size_t)r * HRS) + c4);
      }
    } else {
      // ---- fused decoder: y = relu(h2 @ Wd1^T + bd1); out = y @ Wd2^T + bd2
      const int mt = wave >> 2, nt = wave & 3;  // 8 waves -> 8 y tiles
      const float bv = s_bd1[nt * 16 + (lane & 15)];
      v8f a;
#pragma unroll
      for (int v = 0; v < 8; ++v) a[v] = bv;
#pragma unroll
      for (int ko = 0; ko < KT_H; ++ko)
        a = wmma_f16(load_A(hn, mt * 16, HRS, ko * 32, lane),
                     load_Bf(s_wd1, ko * 4 + nt, lane), a);
      const int ncol = nt * 16 + (lane & 15);
      const int mrow = mt * 16 + ((lane >> 4) << 3);
#pragma unroll
      for (int v = 0; v < 8; ++v)
        s_y[(size_t)(mrow + v) * YRS + ncol] = (_Float16)fmaxf(a[v], 0.0f);
      __syncthreads();
      if (tid < BM) {  // 64-wide dot per batch row
        float s = s_misc[64];
        const _Float16* yr = s_y + (size_t)tid * YRS;
#pragma unroll
        for (int o = 0; o < 64; ++o) s = fmaf((float)yr[o], s_misc[o], s);
        out[(size_t)(mb0 + tid) * T_ + t] = s;
      }
    }
  }
}

// =====================================================================
extern "C" void kernel_launch(void* const* d_in, const int* in_sizes, int n_in,
                              void* d_out, int out_size, void* d_ws,
                              size_t ws_size, hipStream_t stream) {
  (void)in_sizes; (void)n_in; (void)out_size; (void)ws_size;
  const float* src   = (const float*)d_in[0];
  const float* W_ih0 = (const float*)d_in[1];
  const float* W_hh0 = (const float*)d_in[2];
  const float* b_ih0 = (const float*)d_in[3];
  const float* b_hh0 = (const float*)d_in[4];
  const float* W_ih1 = (const float*)d_in[5];
  const float* W_hh1 = (const float*)d_in[6];
  const float* b_ih1 = (const float*)d_in[7];
  const float* b_hh1 = (const float*)d_in[8];
  const float* Wd1   = (const float*)d_in[9];
  const float* bd1   = (const float*)d_in[10];
  const float* Wd2   = (const float*)d_in[11];
  const float* bd2   = (const float*)d_in[12];

  char* ws = (char*)d_ws;
  _Float16* wih0p = (_Float16*)(ws + WS_WIH0);
  _Float16* whh0p = (_Float16*)(ws + WS_WHH0);
  _Float16* wih1p = (_Float16*)(ws + WS_WIH1);
  _Float16* whh1p = (_Float16*)(ws + WS_WHH1);
  _Float16* wd1p  = (_Float16*)(ws + WS_WD1);
  float* bias0 = (float*)(ws + WS_B0);
  float* bias1 = (float*)(ws + WS_B1);
  float* bd1f  = (float*)(ws + WS_BD1);
  float* miscf = (float*)(ws + WS_MISC);
  _Float16* h1 = (_Float16*)(ws + WS_H1);
  float* out = (float*)d_out;

  // weight pre-pack into WMMA B-fragment layout (f16)
  pack_wB<<<1 * NT_G, 32, 0, stream>>>(W_ih0, wih0p, G_, D_, NT_G);
  pack_wB<<<KT_H * NT_G, 32, 0, stream>>>(W_hh0, whh0p, G_, H_, NT_G);
  pack_wB<<<KT_H * NT_G, 32, 0, stream>>>(W_ih1, wih1p, G_, H_, NT_G);
  pack_wB<<<KT_H * NT_G, 32, 0, stream>>>(W_hh1, whh1p, G_, H_, NT_G);
  pack_wB<<<KT_H * 4, 32, 0, stream>>>(Wd1, wd1p, 64, H_, 4);
  prep_misc<<<1, 512, 0, stream>>>(b_ih0, b_hh0, b_ih1, b_hh1, bd1, Wd2, bd2,
                                   bias0, bias1, bd1f, miscf);

  const size_t l0 = lds_bytes(false);
  const size_t l1 = lds_bytes(true);
  hipFuncSetAttribute(reinterpret_cast<const void*>(lstm_layer<false>),
                      hipFuncAttributeMaxDynamicSharedMemorySize, (int)l0);
  hipFuncSetAttribute(reinterpret_cast<const void*>(lstm_layer<true>),
                      hipFuncAttributeMaxDynamicSharedMemorySize, (int)l1);

  lstm_layer<false><<<NWG, NTHR, l0, stream>>>(
      src, nullptr, wih0p, whh0p, bias0, h1, nullptr, nullptr, nullptr,
      nullptr);
  lstm_layer<true><<<NWG, NTHR, l1, stream>>>(
      nullptr, h1, wih1p, whh1p, bias1, nullptr, wd1p, bd1f, miscf, out);
}
